// DMIEstimator_64630667870813
// MI455X (gfx1250) — matrix operand order repaired and checked by reading, not compiled, not run on hardware
//
#include <hip/hip_runtime.h>
#include <hip/hip_fp16.h>
#include <math.h>

typedef __attribute__((ext_vector_type(16))) _Float16 v16h;
typedef __attribute__((ext_vector_type(8)))  _Float16 v8h;
typedef __attribute__((ext_vector_type(8)))  float    v8f;

#define NN   32
#define CC   512
#define HWSZ 196
#define NHW  (NN*HWSZ)      // 6272
#define DD   2048
#define HID  256

__device__ __forceinline__ v8f wmma16(v16h a, v16h b, v8f c) {
  return __builtin_amdgcn_wmma_f32_16x16x32_f16(false, a, false, b, (short)0, c, false, false);
}

// ---- lf NCHW -> (nhw, C) row-major f16, via LDS tile transpose ----
__global__ void k_lf_pack(const float* __restrict__ lf, _Float16* __restrict__ lf16) {
  __shared__ float tile[16][17];
  int n = blockIdx.z, c0 = blockIdx.x * 16, p0 = blockIdx.y * 16;
  int tx = threadIdx.x, ty = threadIdx.y;
  int p = p0 + tx, c = c0 + ty;
  float v = 0.f;
  if (p < HWSZ) v = lf[(n*CC + c)*HWSZ + p];
  tile[ty][tx] = v;
  __syncthreads();
  int pw = p0 + ty, cw = c0 + tx;
  if (pw < HWSZ) lf16[(size_t)(n*HWSZ + pw)*CC + cw] = (_Float16)tile[tx][ty];
}

// ---- pack W1[D:] (512x256) into WMMA B-fragment order, f16 ----
// slot s = kc*16+nt (kc<16,nt<16); lane l; element t: K = kc*32 + 16*(l>>4) + t, N = nt*16 + (l&15)
__global__ void k_w1_pack(const float* __restrict__ W1, _Float16* __restrict__ w1f) {
  int g = blockIdx.x * blockDim.x + threadIdx.x;   // 8192 threads
  int s = g >> 5, l = g & 31;
  int kc = s >> 4, nt = s & 15;
  int col = nt*16 + (l & 15);
  int kr  = kc*32 + (l >> 4)*16;
  _Float16* o = w1f + (size_t)s*512 + l*16;
  #pragma unroll
  for (int t = 0; t < 16; ++t)
    o[t] = (_Float16)W1[(size_t)(DD + kr + t)*HID + col];
}

// ---- pack W2 (256x256) into B-fragment order, f16 ----
__global__ void k_w2_pack(const float* __restrict__ W2, _Float16* __restrict__ w2f) {
  int g = blockIdx.x * blockDim.x + threadIdx.x;   // 4096 threads
  int s = g >> 5, l = g & 31;
  int kc = s >> 4, nt = s & 15;                    // kc<8
  int col = nt*16 + (l & 15);
  int kr  = kc*32 + (l >> 4)*16;
  _Float16* o = w2f + (size_t)s*512 + l*16;
  #pragma unroll
  for (int t = 0; t < 16; ++t)
    o[t] = (_Float16)W2[(size_t)(kr + t)*HID + col];
}

// ---- h_glb = gf @ W1[:D] + b1  (32x256, tiny; fp32) ----
__global__ void k_hglb(const float* __restrict__ gf, const float* __restrict__ W1,
                       const float* __restrict__ b1, float* __restrict__ hglb) {
  int n = blockIdx.x, h = threadIdx.x;
  const float* g = gf + (size_t)n*DD;
  float acc = b1[h];
  #pragma unroll 4
  for (int d = 0; d < DD; ++d) acc = fmaf(g[d], W1[(size_t)d*HID + h], acc);
  hglb[n*HID + h] = acc;
}

// ---- GEMM1: h_loc (6272x256 f32) = lf16 (6272x512) @ W1loc (512x256) ----
__global__ void __launch_bounds__(256) k_hloc(const _Float16* __restrict__ lf16,
                                              const _Float16* __restrict__ w1f,
                                              float* __restrict__ hloc) {
  __shared__ __align__(32) _Float16 Af[2][16*512];   // 2 M-tiles x 16 k-chunks, fragment order
  int j0 = blockIdx.x * 32;
  int tid = threadIdx.x;
  // stage A fragments: lane l<16 holds K {0..15}, l>=16 holds K {16..31} of its chunk
  for (int q = tid; q < 1024; q += 256) {
    int mt = q >> 9, r = q & 511;
    int kc = r >> 5, l = r & 31;
    int m  = mt*16 + (l & 15);
    int kb = (l >> 4) * 16;
    const _Float16* row = lf16 + (size_t)(j0 + m)*CC + kc*32 + kb;
    *(v8h*)&Af[mt][r*16]     = *(const v8h*)row;
    *(v8h*)&Af[mt][r*16 + 8] = *(const v8h*)(row + 8);
  }
  __syncthreads();
  int w = tid >> 5, l = tid & 31;
  int nt0 = 2*w, nt1 = nt0 + 1;
  v8f a00 = {}, a01 = {}, a10 = {}, a11 = {};
  // unroll 2: keep ~2 k-steps of operands in flight, stay under 256 VGPRs
  #pragma unroll 2
  for (int kc = 0; kc < 16; ++kc) {
    v16h A0 = *(const v16h*)&Af[0][(kc*32 + l)*16];
    v16h A1 = *(const v16h*)&Af[1][(kc*32 + l)*16];
    v16h B0 = *(const v16h*)&w1f[(size_t)(kc*16 + nt0)*512 + l*16];
    v16h B1 = *(const v16h*)&w1f[(size_t)(kc*16 + nt1)*512 + l*16];
    a00 = wmma16(A0, B0, a00);  a01 = wmma16(A0, B1, a01);
    a10 = wmma16(A1, B0, a10);  a11 = wmma16(A1, B1, a11);
  }
  int mr = (l >> 4)*8;
  int c0 = nt0*16 + (l & 15), c1 = c0 + 16;
  #pragma unroll
  for (int r = 0; r < 8; ++r) {
    hloc[(size_t)(j0 + mr + r)*HID + c0]      = a00[r];
    hloc[(size_t)(j0 + mr + r)*HID + c1]      = a01[r];
    hloc[(size_t)(j0 + 16 + mr + r)*HID + c0] = a10[r];
    hloc[(size_t)(j0 + 16 + mr + r)*HID + c1] = a11[r];
  }
}

// ---- fused: h1=relu(hglb+hloc); h2=relu(h1@W2+b2); scores=h2@W3+b3 ----
__global__ void __launch_bounds__(256) k_fused(const float* __restrict__ hloc,
    const float* __restrict__ hglb, const _Float16* __restrict__ w2f,
    const float* __restrict__ b2, const float* __restrict__ W3,
    const float* __restrict__ b3, float* __restrict__ scores) {
  __shared__ __align__(32) _Float16 Af[2][8*512];   // 16 KB
  __shared__ float h2s[32][HID + 4];                // padded vs bank conflicts
  __shared__ float hg[HID];
  __shared__ float ps[32][8];
  int jt = blockIdx.x, n = blockIdx.y;
  int j0 = jt * 32;
  int tid = threadIdx.x;
  hg[tid] = hglb[n*HID + tid];
  __syncthreads();
  // stage A = relu(hglb + hloc tile) in f16 fragment order
  for (int q = tid; q < 512; q += 256) {
    int mt = q >> 8, r = q & 255;
    int kc = r >> 5, l = r & 31;
    int m  = mt*16 + (l & 15);
    int kb = (l >> 4) * 16;
    int k0 = kc*32 + kb;
    const float* row = hloc + (size_t)(j0 + m)*HID;
    v8h o0, o1;
    #pragma unroll
    for (int t = 0; t < 8; ++t) {
      o0[t] = (_Float16)fmaxf(row[k0 + t]      + hg[k0 + t],      0.f);
      o1[t] = (_Float16)fmaxf(row[k0 + 8 + t] + hg[k0 + 8 + t], 0.f);
    }
    *(v8h*)&Af[mt][r*16]     = o0;
    *(v8h*)&Af[mt][r*16 + 8] = o1;
  }
  __syncthreads();
  int w = tid >> 5, l = tid & 31;
  int nt0 = 2*w, nt1 = nt0 + 1;
  v8f a00 = {}, a01 = {}, a10 = {}, a11 = {};
  // unroll 2: double-buffer operands without spilling into VGPR-MSB territory
  #pragma unroll 2
  for (int kc = 0; kc < 8; ++kc) {
    v16h A0 = *(const v16h*)&Af[0][(kc*32 + l)*16];
    v16h A1 = *(const v16h*)&Af[1][(kc*32 + l)*16];
    const _Float16* bp0 = &w2f[(size_t)(kc*16 + nt0)*512 + l*16];
    const _Float16* bp1 = &w2f[(size_t)(kc*16 + nt1)*512 + l*16];
    v16h B0 = *(const v16h*)bp0;
    v16h B1 = *(const v16h*)bp1;
    a00 = wmma16(A0, B0, a00);  a01 = wmma16(A0, B1, a01);
    a10 = wmma16(A1, B0, a10);  a11 = wmma16(A1, B1, a11);
  }
  int mr = (l >> 4)*8;
  int c0 = nt0*16 + (l & 15), c1 = c0 + 16;
  float bv0 = b2[c0], bv1 = b2[c1];
  #pragma unroll
  for (int r = 0; r < 8; ++r) {
    h2s[mr + r][c0]      = fmaxf(a00[r] + bv0, 0.f);
    h2s[mr + r][c1]      = fmaxf(a01[r] + bv1, 0.f);
    h2s[16 + mr + r][c0] = fmaxf(a10[r] + bv0, 0.f);
    h2s[16 + mr + r][c1] = fmaxf(a11[r] + bv1, 0.f);
  }
  __syncthreads();
  // scores: deterministic partial-dot + tree reduce (no atomics)
  {
    int m = tid & 31, ch = tid >> 5;        // 8 chunks of 32
    float p = 0.f;
    #pragma unroll
    for (int t = 0; t < 32; ++t) {
      int h = ch*32 + t;
      p = fmaf(h2s[m][h], W3[h], p);
    }
    ps[m][ch] = p;
  }
  __syncthreads();
  if (tid < 32) {
    float s = 0.f;
    #pragma unroll
    for (int ch = 0; ch < 8; ++ch) s += ps[tid][ch];
    scores[(size_t)n*NHW + j0 + tid] = s + b3[0];
  }
}

// ---- contrastive reduction -> mi (32,14,14) ----
__global__ void __launch_bounds__(256) k_final(const float* __restrict__ scores,
                                               float* __restrict__ out) {
  __shared__ float red[256];
  int n = blockIdx.x, tid = threadIdx.x;
  const float* s = scores + (size_t)n*NHW;
  float m = -3.4e38f;
  for (int j = tid; j < NHW; j += 256) m = fmaxf(m, s[j]);
  red[tid] = m; __syncthreads();
  for (int st = 128; st > 0; st >>= 1) {
    if (tid < st) red[tid] = fmaxf(red[tid], red[tid + st]);
    __syncthreads();
  }
  float mx = red[0];
  __syncthreads();
  float acc = 0.f;
  for (int j = tid; j < NHW; j += 256) {
    float e = expf(s[j] - mx);
    if ((j / HWSZ) != n) acc += e;
  }
  red[tid] = acc; __syncthreads();
  for (int st = 128; st > 0; st >>= 1) {
    if (tid < st) red[tid] += red[tid + st];
    __syncthreads();
  }
  float lg = logf(red[0] / (float)(NHW - HWSZ) + 1e-10f);
  for (int k = tid; k < HWSZ; k += 256)
    out[n*HWSZ + k] = s[n*HWSZ + k] - mx - lg;
}

extern "C" void kernel_launch(void* const* d_in, const int* in_sizes, int n_in,
                              void* d_out, int out_size, void* d_ws, size_t ws_size,
                              hipStream_t stream) {
  const float* lf = (const float*)d_in[0];   // (32,512,14,14)
  const float* gf = (const float*)d_in[1];   // (32,2048,1,1)
  const float* W1 = (const float*)d_in[2];   // (2560,256)
  const float* b1 = (const float*)d_in[3];   // (256)
  const float* W2 = (const float*)d_in[4];   // (256,256)
  const float* b2 = (const float*)d_in[5];   // (256)
  const float* W3 = (const float*)d_in[6];   // (256,1)
  const float* b3 = (const float*)d_in[7];   // (1)
  float* out = (float*)d_out;                // (32,14,14)

  char* ws = (char*)d_ws;
  size_t o = 0;
  _Float16* lf16  = (_Float16*)(ws + o); o += (size_t)NHW*CC*2;   // 6.4 MB
  _Float16* w1f   = (_Float16*)(ws + o); o += (size_t)CC*HID*2;   // 256 KB
  _Float16* w2f   = (_Float16*)(ws + o); o += (size_t)HID*HID*2;  // 128 KB
  float*    hglb  = (float*)  (ws + o); o += (size_t)NN*HID*4;    // 32 KB
  float*    hloc  = (float*)  (ws + o); o += (size_t)NHW*HID*4;   // 6.4 MB
  float*    sc    = (float*)  (ws + o); o += (size_t)NN*NHW*4;    // 0.8 MB

  k_lf_pack<<<dim3(CC/16, (HWSZ+15)/16, NN), dim3(16,16), 0, stream>>>(lf, lf16);
  k_w1_pack<<<32, 256, 0, stream>>>(W1, w1f);
  k_w2_pack<<<16, 256, 0, stream>>>(W2, w2f);
  k_hglb<<<NN, HID, 0, stream>>>(gf, W1, b1, hglb);
  k_hloc<<<NHW/32, 256, 0, stream>>>(lf16, w1f, hloc);
  k_fused<<<dim3(NHW/32, NN), 256, 0, stream>>>(hloc, hglb, w2f, b2, W3, b3, sc);
  k_final<<<NN, 256, 0, stream>>>(sc, out);
}
